// Batch_CL_9689446219883
// MI455X (gfx1250) — matrix operand order repaired and checked by reading, not compiled, not run on hardware
//
#include <hip/hip_runtime.h>
#include <stdint.h>

// ---------------------------------------------------------------------------
// Fused supervised-contrastive loss, MI455X (gfx1250, wave32, WMMA).
//   loss = -(1/n_valid) * sum_i [ P_i/C_i - log Z_i ]
// Streaming over columns: logits matrix (256MB) never materialized.
// ---------------------------------------------------------------------------

typedef _Float16 v16h __attribute__((ext_vector_type(16)));
typedef float    v8f  __attribute__((ext_vector_type(8)));
typedef int      v4i  __attribute__((ext_vector_type(4)));

#define N_ROWS 8192
#define DIM    128
#define INV_T  10.0f

#define RPB    128                 // rows per block  (8 waves x 16)
#define NCB    4                   // column blocks (grid.y)
#define CPB    (N_ROWS / NCB)      // 2048 columns per column-block
#define STAGE  64                  // columns per LDS stage (16 KB)
#define NSTAGE (CPB / STAGE)       // 32

// ---- CDNA5 async global->LDS path (guarded; plain copy fallback) ----------
// Probe-confirmed signature (from round-1 diagnostic):
//   void __builtin_amdgcn_global_load_async_to_lds_b128(
//       int4 addrspace(1)* src, int4 addrspace(3)* dst, imm int off, imm cpol)
#if defined(__gfx1250__) && __has_builtin(__builtin_amdgcn_global_load_async_to_lds_b128)
#define CL_ASYNC 1
#define GASYNC_B128(g, l, off)                                                 \
  __builtin_amdgcn_global_load_async_to_lds_b128(                              \
      (__attribute__((address_space(1))) v4i*)(                                \
          (__attribute__((address_space(1))) void*)(void*)(g)),                \
      (__attribute__((address_space(3))) v4i*)(                                \
          (__attribute__((address_space(3))) void*)(void*)(l)),                \
      (off), 0)
__device__ __forceinline__ void wait_async() {
#if __has_builtin(__builtin_amdgcn_s_wait_asynccnt)
  __builtin_amdgcn_s_wait_asynccnt(0);
#else
  asm volatile("s_wait_asynccnt 0" ::: "memory");
#endif
}
#else
#define CL_ASYNC 0
__device__ __forceinline__ void wait_async() {}
#endif

// ---------------------------------------------------------------------------
// Kernel 1: L2-normalize rows, fp32 -> fp16.  One wave32 per row.
// ---------------------------------------------------------------------------
__global__ __launch_bounds__(256) void k_normalize(const float* __restrict__ x,
                                                   _Float16* __restrict__ xh) {
  const int wave = (int)((blockIdx.x * blockDim.x + threadIdx.x) >> 5);
  const int lane = (int)(threadIdx.x & 31);
  if (wave >= N_ROWS) return;

  const float4 v = *(const float4*)(x + (size_t)wave * DIM + lane * 4);
  float ss = v.x * v.x + v.y * v.y + v.z * v.z + v.w * v.w;
#pragma unroll
  for (int m = 1; m < 32; m <<= 1) ss += __shfl_xor(ss, m, 32);
  const float inv = 1.0f / sqrtf(ss);

  union { _Float16 h[4]; uint2 u; } o;
  o.h[0] = (_Float16)(v.x * inv);
  o.h[1] = (_Float16)(v.y * inv);
  o.h[2] = (_Float16)(v.z * inv);
  o.h[3] = (_Float16)(v.w * inv);
  *(uint2*)(xh + (size_t)wave * DIM + lane * 4) = o.u;
}

// ---------------------------------------------------------------------------
// Kernel 2: fused x*x^T + streaming softmax statistics.
//   grid = (N/RPB, NCB); block = 256 threads = 8 waves; wave w owns rows
//   [rb + 16w, rb + 16w + 16).  Columns streamed via double-buffered LDS.
// ---------------------------------------------------------------------------
__global__ __launch_bounds__(256) void k_contrastive(
    const _Float16* __restrict__ xh, const int* __restrict__ labels,
    float* __restrict__ partZ, float* __restrict__ partP,
    float* __restrict__ partC) {
  __shared__ _Float16 lbuf[2][STAGE * DIM];  // 2 x 16 KB

  const int tid  = (int)threadIdx.x;
  const int lane = tid & 31;
  const int wave = tid >> 5;
  const int half = lane >> 4;   // 0: lanes 0-15, 1: lanes 16-31
  const int l16  = lane & 15;

  const int rb = (int)blockIdx.x * RPB;
  const int cb = (int)blockIdx.y * CPB;
  const int mb = rb + wave * 16;           // this wave's first row

  // ---- A fragments: 16 rows x 128 K in ISA layout (4 chunks of K=32) ----
  // lanes 0-15: row M=l16, K = {k0..k0+7, k0+16..k0+23}
  // lanes 16-31: row M=l16, K = {k0+8..k0+15, k0+24..k0+31}
  const _Float16* aptr = xh + (size_t)(mb + l16) * DIM;
  v16h A[4];
#pragma unroll
  for (int kc = 0; kc < 4; ++kc) {
    const int k0 = kc * 32 + (half ? 8 : 0);
    ((uint4*)&A[kc])[0] = *(const uint4*)(aptr + k0);
    ((uint4*)&A[kc])[1] = *(const uint4*)(aptr + k0 + 16);
  }

  // C layout: VGPR r holds (M = r + 8*half, N = l16) -> 8 rows per lane.
  const int growbase = mb + half * 8;
  int rowlab[8];
#pragma unroll
  for (int r = 0; r < 8; ++r) rowlab[r] = labels[growbase + r];

  float accZ[8], accP[8], accC[8];
#pragma unroll
  for (int r = 0; r < 8; ++r) { accZ[r] = 0.f; accP[r] = 0.f; accC[r] = 0.f; }

  // Stage copy: 64 cols x 128 K fp16 is one contiguous 16KB slice of xh.
  auto issue_stage = [&](int s, int b) {
    const _Float16* src = xh + (size_t)(cb + s * STAGE) * DIM + tid * 32;
    _Float16* dst = &lbuf[b][tid * 32];  // 64 bytes per thread
#if CL_ASYNC
    GASYNC_B128(src, dst, 0);
    GASYNC_B128(src, dst, 16);
    GASYNC_B128(src, dst, 32);
    GASYNC_B128(src, dst, 48);
#else
    const uint4* s4 = (const uint4*)src;
    uint4* d4 = (uint4*)dst;
    d4[0] = s4[0]; d4[1] = s4[1]; d4[2] = s4[2]; d4[3] = s4[3];
#endif
  };

  issue_stage(0, 0);
  wait_async();
  __syncthreads();

  int buf = 0;
  for (int s = 0; s < NSTAGE; ++s) {
    if (s + 1 < NSTAGE) issue_stage(s + 1, buf ^ 1);  // prefetch next stage

#pragma unroll
    for (int t = 0; t < STAGE / 16; ++t) {
      const int gcol   = cb + s * STAGE + t * 16 + l16;
      const int collab = labels[gcol];

      // B fragment: column N=l16 of tile; lanes 0-15 hold K {k0..k0+7, k0+16..k0+23},
      // lanes 16-31 hold K {k0+8..k0+15, k0+24..k0+31} (same as A).
      const _Float16* bp = &lbuf[buf][(t * 16 + l16) * DIM];
      v16h B[4];
#pragma unroll
      for (int kc = 0; kc < 4; ++kc) {
        const int k0 = kc * 32 + half * 8;
        ((uint4*)&B[kc])[0] = *(const uint4*)(bp + k0);
        ((uint4*)&B[kc])[1] = *(const uint4*)(bp + k0 + 16);
      }

      v8f c = {0.f, 0.f, 0.f, 0.f, 0.f, 0.f, 0.f, 0.f};
#pragma unroll
      for (int kc = 0; kc < 4; ++kc)
        c = __builtin_amdgcn_wmma_f32_16x16x32_f16(false, A[kc], false, B[kc],
                                                   (short)0, c, false, false);

#pragma unroll
      for (int r = 0; r < 8; ++r) {
        const float logit = c[r] * INV_T;
        const float e = __expf(logit);
        const bool diag = (gcol == growbase + r);
        const bool pos  = (collab == rowlab[r]) && !diag;
        accZ[r] += diag ? 0.0f : e;
        accP[r] += pos ? logit : 0.0f;
        accC[r] += pos ? 1.0f : 0.0f;
      }
    }

    wait_async();       // own async writes to next buffer landed in LDS
    __syncthreads();    // everyone's writes visible; prev buffer free
    buf ^= 1;
  }

  // Half-wave reduction (xor masks < 16 keep the two halves separate).
#pragma unroll
  for (int r = 0; r < 8; ++r) {
    float z = accZ[r], p = accP[r], cn = accC[r];
#pragma unroll
    for (int m = 1; m < 16; m <<= 1) {
      z  += __shfl_xor(z, m, 32);
      p  += __shfl_xor(p, m, 32);
      cn += __shfl_xor(cn, m, 32);
    }
    if (l16 == 0) {
      const size_t idx = (size_t)blockIdx.y * N_ROWS + (growbase + r);
      partZ[idx] = z; partP[idx] = p; partC[idx] = cn;
    }
  }
}

// ---------------------------------------------------------------------------
// Kernel 3: fold column partials, per-row term, reduce to scalar loss.
// ---------------------------------------------------------------------------
__global__ __launch_bounds__(256) void k_finalize(
    const float* __restrict__ partZ, const float* __restrict__ partP,
    const float* __restrict__ partC, float* __restrict__ out) {
  __shared__ float sT[256], sV[256];
  const int tid = (int)threadIdx.x;
  float sumT = 0.f, sumV = 0.f;
  for (int row = tid; row < N_ROWS; row += 256) {
    float Z = 0.f, P = 0.f, C = 0.f;
#pragma unroll
    for (int c = 0; c < NCB; ++c) {
      Z += partZ[c * N_ROWS + row];
      P += partP[c * N_ROWS + row];
      C += partC[c * N_ROWS + row];
    }
    if (C > 0.5f) { sumT += P / C - logf(Z); sumV += 1.0f; }
  }
  sT[tid] = sumT; sV[tid] = sumV;
  __syncthreads();
  for (int off = 128; off > 0; off >>= 1) {
    if (tid < off) { sT[tid] += sT[tid + off]; sV[tid] += sV[tid + off]; }
    __syncthreads();
  }
  if (tid == 0) out[0] = -sT[0] / sV[0];
}

// ---------------------------------------------------------------------------
extern "C" void kernel_launch(void* const* d_in, const int* in_sizes, int n_in,
                              void* d_out, int out_size, void* d_ws,
                              size_t ws_size, hipStream_t stream) {
  const float* x      = (const float*)d_in[0];   // feature_embeds [8192,128] f32
  const int*   labels = (const int*)d_in[1];     // label_ids [8192]
  float* out = (float*)d_out;

  char* ws = (char*)d_ws;
  _Float16* xh  = (_Float16*)ws;                               // 2 MB
  float* partZ  = (float*)(ws + (size_t)N_ROWS * DIM * 2);     // NCB*8192 f32
  float* partP  = partZ + (size_t)NCB * N_ROWS;
  float* partC  = partP + (size_t)NCB * N_ROWS;

  k_normalize<<<N_ROWS / 8, 256, 0, stream>>>(x, xh);

  dim3 g2(N_ROWS / RPB, NCB);
  k_contrastive<<<g2, 256, 0, stream>>>(xh, labels, partZ, partP, partC);

  k_finalize<<<1, 256, 0, stream>>>(partZ, partP, partC, out);
}
